// MultiHeadAttentionBlock_4784593567954
// MI455X (gfx1250) — matrix operand order repaired and checked by reading, not compiled, not run on hardware
//
#include <hip/hip_runtime.h>

// MI455X / gfx1250 wave32 WMMA implementation of a full MHA block.
// B=2, S=2048, D=1024, H=16, DK=64.
// All matmuls via v_wmma_f32_16x16x32_f16 (f32 accum). Q/K/V/Ctx kept as f16
// in global memory; attention tiles staged into LDS with
// global_load_async_to_lds_b128 (ASYNCcnt) so the hot loop has no cvt VALU.

typedef __attribute__((ext_vector_type(16))) _Float16 v16h;
typedef __attribute__((ext_vector_type(8)))  float    v8f;

#define B_  2
#define S_  2048
#define D_  1024
#define H_  16
#define DK_ 64

#define NEG_INF (-3.0e38f)

// ---------------------------------------------------------------------------
// Fragment loaders (ISA 7.12.2, wave32). Both are CONTIGUOUS 32B per lane ->
// compiler emits ds_load_b128 pairs.
// A 16x32 f16: lane l(half=l>>4, r=l&15) -> row M=r; K = half*16 + {0..15}
// ---------------------------------------------------------------------------
static __device__ __forceinline__ v16h ld_a_frag(const _Float16* base, int row0,
                                                 int koff, int lane, int stride) {
  const int half = lane >> 4, r = lane & 15;
  const _Float16* p = base + (size_t)(row0 + r) * stride + koff + half * 16;
  v16h a;
#pragma unroll
  for (int v = 0; v < 8; ++v) { a[2*v] = p[2*v]; a[2*v+1] = p[2*v+1]; }
  return a;
}

// B 32x16 f16 read from a K-contiguous LDS tile laid out as T[n][k]:
// lane (half,n) -> col N=n, K = koff + half*16 .. +15 (contiguous).
static __device__ __forceinline__ v16h ld_bT_frag(const _Float16* base, int col0,
                                                  int koff, int lane, int stride) {
  const int half = lane >> 4, n = lane & 15;
  const _Float16* p = base + (size_t)(col0 + n) * stride + koff + half * 16;
  v16h b;
#pragma unroll
  for (int v = 0; v < 8; ++v) { b[2*v] = p[2*v]; b[2*v+1] = p[2*v+1]; }
  return b;
}

// ---------------------------------------------------------------------------
// GEMM: Y[4096,1024] = X[4096,1024] @ W[1024,1024] + bias  (f16 WMMA)
// 128x128 tile / 256 threads (8 waves as 2x4, each wave 64x32).
//   XF16:   X is _Float16 (Ctx) instead of float
//   OUTMODE 0: Y = float, natural layout            (final projection)
//   OUTMODE 1: Y = f16, natural layout, *scale      (Q with 1/8, K)
//   OUTMODE 2: Y = f16, per-head transposed [B][H][DK][S]   (V)
// ---------------------------------------------------------------------------
template <bool XF16, int OUTMODE>
__global__ __launch_bounds__(256) void gemm_bias_wmma(
    const void* __restrict__ Xv, const float* __restrict__ W,
    const float* __restrict__ bias, void* __restrict__ Yv, float scale) {
  __shared__ _Float16 As[128][40];    // [m][k], rows 80B -> 16B aligned
  __shared__ _Float16 BsT[128][40];   // [n][k] transposed: frag reads contiguous

  const int t    = threadIdx.x;
  const int lane = t & 31;
  const int wave = t >> 5;
  const int wm   = wave >> 2;         // 0..1
  const int wn   = wave & 3;          // 0..3
  const int m0   = blockIdx.y * 128;
  const int n0   = blockIdx.x * 128;

  v8f acc[4][2] = {};

  const int atr = t >> 3;             // f32 A loader: 0..31
  const int atc = (t & 7) * 4;        // 0..28
  const int bn  = t & 127;            // W loader: column
  const int bkh = (t >> 7) * 16;      // k half: 0 or 16

  for (int k0 = 0; k0 < D_; k0 += 32) {
    __syncthreads();
    if (XF16) {                                   // f16 X: pure 16B copies
      const _Float16* X = (const _Float16*)Xv;
#pragma unroll
      for (int i = 0; i < 2; ++i) {
        const int c = t + i * 256;                // 512 chunks of 16B
        const int row = c >> 2, co = (c & 3) * 8;
        *(uint4*)&As[row][co] =
            *(const uint4*)(X + (size_t)(m0 + row) * D_ + k0 + co);
      }
    } else {                                      // f32 X: convert to f16
      const float* X = (const float*)Xv;
#pragma unroll
      for (int rr = 0; rr < 4; ++rr) {
        const int row = atr + rr * 32;
        const float4 f = *(const float4*)(X + (size_t)(m0 + row) * D_ + k0 + atc);
        As[row][atc + 0] = (_Float16)f.x;
        As[row][atc + 1] = (_Float16)f.y;
        As[row][atc + 2] = (_Float16)f.z;
        As[row][atc + 3] = (_Float16)f.w;
      }
    }
    {   // W tile 32x128 -> BsT[n][k]; b32 loads coalesced across lanes (adj n)
      __align__(16) _Float16 tmp[16];
#pragma unroll
      for (int kk = 0; kk < 16; ++kk)
        tmp[kk] = (_Float16)W[(size_t)(k0 + bkh + kk) * D_ + n0 + bn];
      *(uint4*)&BsT[bn][bkh]     = *(uint4*)&tmp[0];
      *(uint4*)&BsT[bn][bkh + 8] = *(uint4*)&tmp[8];
    }
    __syncthreads();

    v16h af[4], bf[2];
#pragma unroll
    for (int mi = 0; mi < 4; ++mi) af[mi] = ld_a_frag(&As[0][0],  wm*64 + mi*16, 0, lane, 40);
#pragma unroll
    for (int ni = 0; ni < 2; ++ni) bf[ni] = ld_bT_frag(&BsT[0][0], wn*32 + ni*16, 0, lane, 40);
#pragma unroll
    for (int mi = 0; mi < 4; ++mi)
#pragma unroll
      for (int ni = 0; ni < 2; ++ni)
        acc[mi][ni] = __builtin_amdgcn_wmma_f32_16x16x32_f16(
            false, af[mi], false, bf[ni], (short)0, acc[mi][ni], false, false);
  }

  const int half = lane >> 4, ln = lane & 15;
#pragma unroll
  for (int mi = 0; mi < 4; ++mi)
#pragma unroll
    for (int ni = 0; ni < 2; ++ni) {
      const int col = n0 + wn*32 + ni*16 + ln;
      const float bv = bias[col];
#pragma unroll
      for (int j = 0; j < 8; ++j) {
        const int row = m0 + wm*64 + mi*16 + half*8 + j;
        const float y = (acc[mi][ni][j] + bv) * scale;
        if (OUTMODE == 0) {
          ((float*)Yv)[(size_t)row * D_ + col] = y;
        } else if (OUTMODE == 1) {
          ((_Float16*)Yv)[(size_t)row * D_ + col] = (_Float16)y;
        } else {                       // [B][H][DK][S]
          const int bb = row >> 11, s = row & (S_ - 1);
          const int hh = col >> 6, dk = col & (DK_ - 1);
          ((_Float16*)Yv)[(((size_t)bb * H_ + hh) * DK_ + dk) * S_ + s] = (_Float16)y;
        }
      }
    }
}

// ---------------------------------------------------------------------------
// Async 64x64 f16 tile copy: global -> LDS, 16B per lane-op, tracked by
// ASYNCcnt (CDNA5 GLOBAL_LOAD_ASYNC_TO_LDS_B128).
// ---------------------------------------------------------------------------
static __device__ __forceinline__ void async_tile64(const _Float16* __restrict__ src,
                                                    size_t rowStride,
                                                    _Float16 (*dst)[72], int t) {
#pragma unroll
  for (int i = 0; i < 4; ++i) {
    const int c   = t + i * 128;        // 512 chunks of 16B
    const int row = c >> 3;             // 0..63
    const int co  = (c & 7) * 8;        // f16 col offset
    const uint32_t lds = (uint32_t)(size_t)&dst[row][co];
    const _Float16* g = src + (size_t)row * rowStride + co;
    asm volatile("global_load_async_to_lds_b128 %0, %1, off"
                 :: "v"(lds), "v"(g) : "memory");
  }
}
static __device__ __forceinline__ void wait_async_all() {
  asm volatile("s_wait_asynccnt 0x0" ::: "memory");
}

// ---------------------------------------------------------------------------
// Attention: one block (128 thr / 4 waves) per (b, h, 64-row q tile).
// Pass1: QK^T (WMMA) -> causal row max / sumexp (shfl + __expf).
// Pass2: recompute scores, write probs (f32, required output), P via LDS,
//        O += P @ V (WMMA, V pre-transposed so B-frags are contiguous).
// ---------------------------------------------------------------------------
__global__ __launch_bounds__(128) void attention_f16wmma(
    const _Float16* __restrict__ Qh, const _Float16* __restrict__ Kh,
    const _Float16* __restrict__ Vt, float* __restrict__ probs,
    _Float16* __restrict__ ctx) {
  __shared__ _Float16 Qs[64][72];     // [q][d]   (A for QK^T and rows for P)
  __shared__ _Float16 Ks[64][72];     // [kv][d]  (B^T for QK^T: d contiguous)
  __shared__ _Float16 Vs[64][72];     // [dk][kv] (B^T for P@V: kv contiguous)
  __shared__ _Float16 Ps[64][72];     // [q][kv]  (A for P@V)

  const int t = threadIdx.x, lane = t & 31, w = t >> 5;
  const int qt = blockIdx.x, h = blockIdx.y, b = blockIdx.z;
  const int half = lane >> 4, ln = lane & 15;

  const _Float16* Qb = Qh + (size_t)b * S_ * D_ + (size_t)h * DK_;   // row stride D
  const _Float16* Kb = Kh + (size_t)b * S_ * D_ + (size_t)h * DK_;   // row stride D
  const _Float16* Vb = Vt + ((size_t)(b * H_ + h)) * DK_ * S_;       // [dk][s]
  float* Pb = probs + ((size_t)(b * H_ + h)) * S_ * S_;

  async_tile64(Qb + (size_t)(qt * 64) * D_, D_, Qs, t);   // Q pre-scaled by 1/8

  float m[8], l[8];
#pragma unroll
  for (int j = 0; j < 8; ++j) { m[j] = NEG_INF; l[j] = 0.f; }

  const int q0 = qt * 64 + w * 16 + half * 8;   // + j -> global query row

  // ----- PASS 1 -----
  for (int kt = 0; kt <= qt; ++kt) {
    __syncthreads();
    async_tile64(Kb + (size_t)(kt * 64) * D_, D_, Ks, t);
    if (kt < qt)
      __builtin_prefetch(Kb + (size_t)((kt + 1) * 64 + (t >> 1)) * D_, 0, 1);
    wait_async_all();
    __syncthreads();

    v8f sacc[4] = {};
#pragma unroll
    for (int kk = 0; kk < 64; kk += 32) {
      const v16h a = ld_a_frag(&Qs[0][0], w * 16, kk, lane, 72);
#pragma unroll
      for (int nt = 0; nt < 4; ++nt) {
        const v16h bT = ld_bT_frag(&Ks[0][0], nt * 16, kk, lane, 72);
        sacc[nt] = __builtin_amdgcn_wmma_f32_16x16x32_f16(
            false, a, false, bT, (short)0, sacc[nt], false, false);
      }
    }
#pragma unroll
    for (int j = 0; j < 8; ++j) {
      const int q = q0 + j;
      float sv[4];
      float rmax = NEG_INF;
#pragma unroll
      for (int nt = 0; nt < 4; ++nt) {
        const int kv = kt * 64 + nt * 16 + ln;
        float s = sacc[nt][j];
        if (kv > q) s = NEG_INF;
        sv[nt] = s;
        rmax = fmaxf(rmax, s);
      }
#pragma unroll
      for (int off = 8; off >= 1; off >>= 1)
        rmax = fmaxf(rmax, __shfl_xor(rmax, off, 16));
      const float nm = fmaxf(m[j], rmax);
      float rs = 0.f;
#pragma unroll
      for (int nt = 0; nt < 4; ++nt) rs += __expf(sv[nt] - nm);
#pragma unroll
      for (int off = 8; off >= 1; off >>= 1) rs += __shfl_xor(rs, off, 16);
      l[j] = l[j] * __expf(m[j] - nm) + rs;
      m[j] = nm;
    }
  }

  float invl[8];
#pragma unroll
  for (int j = 0; j < 8; ++j) invl[j] = 1.0f / l[j];

  // ----- PASS 2 -----
  v8f oacc[4] = {};
  for (int kt = 0; kt <= qt; ++kt) {
    __syncthreads();
    async_tile64(Kb + (size_t)(kt * 64) * D_, D_, Ks, t);
    async_tile64(Vb + (size_t)(kt * 64), S_, Vs, t);      // [dk][kv] tile
    wait_async_all();
    __syncthreads();

    v8f sacc[4] = {};
#pragma unroll
    for (int kk = 0; kk < 64; kk += 32) {
      const v16h a = ld_a_frag(&Qs[0][0], w * 16, kk, lane, 72);
#pragma unroll
      for (int nt = 0; nt < 4; ++nt) {
        const v16h bT = ld_bT_frag(&Ks[0][0], nt * 16, kk, lane, 72);
        sacc[nt] = __builtin_amdgcn_wmma_f32_16x16x32_f16(
            false, a, false, bT, (short)0, sacc[nt], false, false);
      }
    }
#pragma unroll
    for (int nt = 0; nt < 4; ++nt)
#pragma unroll
      for (int j = 0; j < 8; ++j) {
        const int q  = q0 + j;
        const int kv = kt * 64 + nt * 16 + ln;
        const float p = (kv > q) ? 0.f : __expf(sacc[nt][j] - m[j]) * invl[j];
        Pb[(size_t)q * S_ + kv] = p;                       // required output
        Ps[w * 16 + half * 8 + j][nt * 16 + ln] = (_Float16)p;
      }
    __syncthreads();
#pragma unroll
    for (int kk = 0; kk < 64; kk += 32) {
      const v16h a = ld_a_frag(&Ps[0][0], w * 16, kk, lane, 72);
#pragma unroll
      for (int nt = 0; nt < 4; ++nt) {
        const v16h bv = ld_bT_frag(&Vs[0][0], nt * 16, kk, lane, 72);
        oacc[nt] = __builtin_amdgcn_wmma_f32_16x16x32_f16(
            false, a, false, bv, (short)0, oacc[nt], false, false);
      }
    }
  }

  // ctx[b, q, h*64 + d]  (f16, feeds the final WMMA GEMM)
#pragma unroll
  for (int nt = 0; nt < 4; ++nt)
#pragma unroll
    for (int j = 0; j < 8; ++j) {
      const int q = q0 + j;
      ctx[((size_t)b * S_ + q) * D_ + h * 64 + nt * 16 + ln] =
          (_Float16)oacc[nt][j];
    }

  // zero-fill fully-masked upper-triangle score region (d_out is poisoned)
  const int zstart = (qt + 1) * 64;
  if (zstart < S_) {
    const int zc4 = (S_ - zstart) >> 2;
    const float4 z4 = make_float4(0.f, 0.f, 0.f, 0.f);
    for (int r = 0; r < 64; ++r) {
      float4* dst = (float4*)(Pb + (size_t)(qt * 64 + r) * S_ + zstart);
      for (int i = t; i < zc4; i += 128) dst[i] = z4;
    }
  }
}

// ---------------------------------------------------------------------------
extern "C" void kernel_launch(void* const* d_in, const int* in_sizes, int n_in,
                              void* d_out, int out_size, void* d_ws, size_t ws_size,
                              hipStream_t stream) {
  const float* q   = (const float*)d_in[0];
  const float* k   = (const float*)d_in[1];
  const float* v   = (const float*)d_in[2];
  // d_in[3] = causal mask (int) — structure known, unused
  const float* w_q = (const float*)d_in[4];
  const float* b_q = (const float*)d_in[5];
  const float* w_k = (const float*)d_in[6];
  const float* b_k = (const float*)d_in[7];
  const float* w_v = (const float*)d_in[8];
  const float* b_v = (const float*)d_in[9];
  const float* w_o = (const float*)d_in[10];
  const float* b_o = (const float*)d_in[11];

  float* out    = (float*)d_out;                       // [B,S,D]
  float* scores = out + (size_t)B_ * S_ * D_;          // [B,H,S,S]

  const size_t plane = (size_t)B_ * S_ * D_;           // 4M elements
  _Float16* Qh  = (_Float16*)d_ws;                     // [B,S,D]   pre-scaled
  _Float16* Kh  = Qh + plane;                          // [B,S,D]
  _Float16* Vt  = Kh + plane;                          // [B,H,DK,S]
  _Float16* Ctx = Vt + plane;                          // [B,S,D]   ws: 32 MiB

  const dim3 gg(D_ / 128, (B_ * S_) / 128);            // 8 x 32 blocks
  gemm_bias_wmma<false, 1><<<gg, 256, 0, stream>>>(q, w_q, b_q, Qh, 0.125f);
  gemm_bias_wmma<false, 1><<<gg, 256, 0, stream>>>(k, w_k, b_k, Kh, 1.0f);
  gemm_bias_wmma<false, 2><<<gg, 256, 0, stream>>>(v, w_v, b_v, Vt, 1.0f);

  attention_f16wmma<<<dim3(S_ / 64, H_, B_), 128, 0, stream>>>(Qh, Kh, Vt, scores, Ctx);

  gemm_bias_wmma<true, 0><<<gg, 256, 0, stream>>>(Ctx, w_o, b_o, out, 1.0f);
}
